// MambaIntegerBlock_6442450944358
// MI455X (gfx1250) — compile-verified, matrix operand order checked by reading, not run on hardware
//
#include <hip/hip_runtime.h>
#include <hip/hip_bf16.h>

#define B_  2
#define L_  1024
#define DM  1024
#define DI  2048
#define DS  16
#define DR  64

typedef __attribute__((ext_vector_type(16))) __bf16 v16bf;
typedef __attribute__((ext_vector_type(8)))  float  v8f;
typedef __attribute__((ext_vector_type(4)))  unsigned int v4u;

// ---------- helpers ----------
__device__ inline unsigned short f32_to_bf16(float f) {
    union { float f; unsigned u; } c; c.f = f;
    unsigned x = c.u;
    unsigned r = x + 0x7FFFu + ((x >> 16) & 1u);   // RNE
    return (unsigned short)(r >> 16);
}
__device__ inline float bf16_to_f32(unsigned short h) {
    union { unsigned u; float f; } c; c.u = ((unsigned)h) << 16; return c.f;
}
__device__ inline float clampf(float x, float lo, float hi) {
    return fminf(fmaxf(x, lo), hi);
}
__device__ inline float pow2_scale(float var) {
    float v = var + 1e-9f;
    float s = 1.0f;
    if (v >= 4.0f)      s = 0.5f;
    if (v >= 16.0f)     s = 0.25f;
    if (v >= 64.0f)     s = 0.125f;
    if (v >= 256.0f)    s = 0.0625f;
    if (v >= 1024.0f)   s = 0.03125f;
    if (v >= 4096.0f)   s = 0.015625f;
    if (v >= 16384.0f)  s = 0.0078125f;
    if (v >= 65536.0f)  s = 0.00390625f;
    if (v < 1.0f)       s = 1.0f;
    if (v < 0.25f)      s = 2.0f;
    if (v < 0.0625f)    s = 4.0f;
    if (v < 0.015625f)  s = 8.0f;
    return s;
}
__device__ inline float rsqrt_newton3(float y) {
    y = clampf(y, 1e-6f, 1e6f);
    float r = (y > 16.0f) ? 0.125f : 0.5f;
    if (y > 64.0f)  r = 0.0625f;
    if (y > 256.0f) r = 0.03125f;
    if (y < 4.0f)   r = 1.0f;
    if (y < 1.0f)   r = 2.0f;
    if (y < 0.25f)  r = 4.0f;
#pragma unroll
    for (int i = 0; i < 3; ++i) {
        r = r * (1.5f - 0.5f * y * r * r);
        r = clampf(r, 1e-6f, 1000.0f);
    }
    return r;
}
__device__ inline float squareplus3(float x) {
    float y  = x * x + 4.0f;
    float ys = clampf(y, 1e-6f, 1e6f);
    float sq = clampf(ys * rsqrt_newton3(ys), 0.0f, 1000.0f);
    return 0.5f * (x + sq);
}

// ---------- kernel 1: bitshift norm -> bf16 ----------
__global__ __launch_bounds__(256)
void norm_kernel(const float* __restrict__ x, const float* __restrict__ gamma,
                 unsigned short* __restrict__ out) {
    __shared__ float sm[256];
    const int tok = blockIdx.x;                 // 0..2047
    const int t   = threadIdx.x;
    const float* row = x + (size_t)tok * DM;
    float v[4];
#pragma unroll
    for (int i = 0; i < 4; ++i) v[i] = row[t + 256 * i];

    float s = v[0] + v[1] + v[2] + v[3];
    sm[t] = s; __syncthreads();
    for (int o = 128; o > 0; o >>= 1) { if (t < o) sm[t] += sm[t + o]; __syncthreads(); }
    float mean1 = sm[0] * (1.0f / DM); __syncthreads();

#pragma unroll
    for (int i = 0; i < 4; ++i) v[i] -= mean1;
    s = v[0] + v[1] + v[2] + v[3];
    sm[t] = s; __syncthreads();
    for (int o = 128; o > 0; o >>= 1) { if (t < o) sm[t] += sm[t + o]; __syncthreads(); }
    float mean2 = sm[0] * (1.0f / DM); __syncthreads();

    float sq = 0.0f;
#pragma unroll
    for (int i = 0; i < 4; ++i) { v[i] -= mean2; sq += v[i] * v[i]; }
    sm[t] = sq; __syncthreads();
    for (int o = 128; o > 0; o >>= 1) { if (t < o) sm[t] += sm[t + o]; __syncthreads(); }
    float var = sm[0] * (1.0f / DM);
    float sc  = pow2_scale(var);

#pragma unroll
    for (int i = 0; i < 4; ++i) {
        int c = t + 256 * i;
        out[(size_t)tok * DM + c] = f32_to_bf16(v[i] * sc * gamma[c]);
    }
}

// ---------- kernel 2: fp32 -> bf16 convert ----------
__global__ __launch_bounds__(256)
void cvt_bf16_kernel(const float* __restrict__ src, unsigned short* __restrict__ dst, int n) {
    int i = blockIdx.x * 256 + threadIdx.x;
    if (i < n) dst[i] = f32_to_bf16(src[i]);
}

// ---------- WMMA fragment load (row-major [rows][K] bf16) ----------
// ISA 7.12.2: lane<16 holds K={k..k+7, k+16..k+23}; lane>=16 holds the +8 halves.
__device__ inline v16bf load_frag_bf16(const unsigned short* __restrict__ base,
                                       int row, int ld, int k0) {
    union { v4u q[2]; v16bf v; } f;
    const unsigned short* p = base + (size_t)row * ld + k0;
    f.q[0] = *reinterpret_cast<const v4u*>(p);        // K = k0 .. k0+7
    f.q[1] = *reinterpret_cast<const v4u*>(p + 16);   // K = k0+16 .. k0+23
    return f.v;
}

// ---------- kernel 3a: register-blocked WMMA GEMM (big GEMMs) ----------
// C[m][n] = sum_k A[m][k]*Bw[n][k].  Block = 4 waves (2Mx2N) -> 128x64 tile;
// wave tile = 64x32 = 4x2 subtiles: 6 fragment loads feed 8 WMMAs per k-step.
// Requires M%128==0, N%64==0, K%32==0.
__global__ __launch_bounds__(128)
void gemm_bf16_wmma_big(const unsigned short* __restrict__ A,
                        const unsigned short* __restrict__ Bw,
                        float* __restrict__ Cout,
                        const float* __restrict__ residual,
                        const float* __restrict__ gate,
                        int M, int N, int K, float clipLo, float clipHi) {
    const int lane = threadIdx.x & 31;
    const int wave = threadIdx.x >> 5;
    const int m0 = blockIdx.y * 128 + (wave >> 1) * 64;
    const int n0 = blockIdx.x * 64  + (wave & 1) * 32;
    const int rla  = lane & 15;
    const int ksel = (lane >> 4) << 3;

    v8f acc[4][2] = {};
    for (int k = 0; k < K; k += 32) {
        v16bf a[4], b[2];
#pragma unroll
        for (int i = 0; i < 4; ++i)
            a[i] = load_frag_bf16(A, m0 + 16 * i + rla, K, k + ksel);
#pragma unroll
        for (int j = 0; j < 2; ++j)
            b[j] = load_frag_bf16(Bw, n0 + 16 * j + rla, K, k + ksel);
#pragma unroll
        for (int i = 0; i < 4; ++i)
#pragma unroll
            for (int j = 0; j < 2; ++j)
                acc[i][j] = __builtin_amdgcn_wmma_f32_16x16x32_bf16(
                    false, a[i], false, b[j], (short)0, acc[i][j], false, false);
    }

    const float gv   = gate ? gate[0] : 1.0f;
    const int mbase  = (lane >> 4) << 3;
    const int col0   = lane & 15;
#pragma unroll
    for (int i = 0; i < 4; ++i)
#pragma unroll
        for (int j = 0; j < 2; ++j)
#pragma unroll
            for (int r = 0; r < 8; ++r) {
                float v = clampf(acc[i][j][r], clipLo, clipHi);
                size_t idx = (size_t)(m0 + 16 * i + mbase + r) * N + (n0 + 16 * j + col0);
                if (residual) v = residual[idx] + v * gv;
                Cout[idx] = v;
            }
}

// ---------- kernel 3b: simple WMMA GEMM (small-N GEMM, N%16==0) ----------
__global__ __launch_bounds__(128)
void gemm_bf16_wmma(const unsigned short* __restrict__ A,
                    const unsigned short* __restrict__ Bw,
                    float* __restrict__ Cout,
                    int M, int N, int K, float clipLo, float clipHi) {
    const int lane = threadIdx.x & 31;
    const int wave = threadIdx.x >> 5;
    const int m0 = blockIdx.y * 16;
    const int n0 = blockIdx.x * 64 + wave * 16;
    if (m0 >= M || n0 >= N) return;

    const int ra   = m0 + (lane & 15);
    const int rb   = n0 + (lane & 15);
    const int ksel = (lane >> 4) << 3;

    v8f acc = {};
    for (int k = 0; k < K; k += 32) {
        v16bf a = load_frag_bf16(A,  ra, K, k + ksel);
        v16bf b = load_frag_bf16(Bw, rb, K, k + ksel);
        acc = __builtin_amdgcn_wmma_f32_16x16x32_bf16(
                  false, a, false, b, (short)0, acc, false, false);
    }

    const int mrow = m0 + ((lane >> 4) << 3);
    const int col  = n0 + (lane & 15);
#pragma unroll
    for (int i = 0; i < 8; ++i) {
        float v = clampf(acc[i], clipLo, clipHi);
        Cout[(size_t)(mrow + i) * N + col] = v;
    }
}

// ---------- kernel 4: causal depthwise conv(4) + squareplus -> bf16 ----------
__global__ __launch_bounds__(256)
void conv_act_kernel(const float* __restrict__ xz,       // [B*L][2*DI], x = cols 0..DI-1
                     const float* __restrict__ conv_w,   // [DI][4]
                     const float* __restrict__ conv_b,
                     unsigned short* __restrict__ xact) {
    int idx = blockIdx.x * 256 + threadIdx.x;
    if (idx >= B_ * L_ * DI) return;
    int d = idx & (DI - 1);
    int l = (idx >> 11) & (L_ - 1);
    int b = idx >> 21;
    const float* xcol = xz + (size_t)(b * L_) * (2 * DI) + d;
    float acc = conv_b[d];
#pragma unroll
    for (int j = 0; j < 4; ++j) {
        int li = l - 3 + j;                    // pad(3,3) then [:L] => taps l-3..l
        if (li >= 0) acc += conv_w[d * 4 + j] * xcol[(size_t)li * (2 * DI)];
    }
    acc = clampf(acc, -50.0f, 50.0f);
    xact[idx] = f32_to_bf16(squareplus3(acc));
}

// ---------- kernel 5: slice dt (= xp[:, :64]) to bf16 for GEMM3 ----------
__global__ __launch_bounds__(256)
void slice_dt_kernel(const float* __restrict__ xp, unsigned short* __restrict__ dt) {
    int i = blockIdx.x * 256 + threadIdx.x;
    if (i >= B_ * L_ * DR) return;
    int m = i / DR, k = i % DR;
    dt[i] = f32_to_bf16(xp[(size_t)m * (DR + 2 * DS) + k]);
}

// ---------- kernel 6: fused dyadic scan + C-contraction + sigmoid gate ----------
// one thread per (b,d): 16 states in registers; B/C vectors staged in LDS per L-chunk.
#define CHUNK 64
__global__ __launch_bounds__(256)
void scan_kernel(const unsigned short* __restrict__ xact,   // [B*L][DI] bf16
                 const float* __restrict__ dmod,            // [B*L][DI]
                 const float* __restrict__ xp,              // [B*L][96]
                 const float* __restrict__ xz,              // [B*L][2*DI] (z = cols DI..)
                 const float* __restrict__ base_decay,      // [DI][DS]
                 const float* __restrict__ decay_shifts,    // [DI][DS]
                 unsigned short* __restrict__ ybf) {        // [B*L][DI] bf16
    __shared__ float sB[CHUNK][DS];
    __shared__ float sC[CHUNK][DS];
    const int gid = blockIdx.x * 256 + threadIdx.x;   // 0..4095
    const int b = gid >> 11;                          // all threads in a block share b
    const int d = gid & (DI - 1);

    float h[DS], basev[DS], rsh[DS];
#pragma unroll
    for (int s = 0; s < DS; ++s) {
        h[s] = 0.0f;
        basev[s] = base_decay[d * DS + s];
        rsh[s]   = exp2f(-decay_shifts[d * DS + s]);
    }

    for (int l0 = 0; l0 < L_; l0 += CHUNK) {
        __syncthreads();
        for (int i = threadIdx.x; i < CHUNK * DS; i += 256) {
            int ll = i >> 4, s = i & (DS - 1);
            size_t r = (size_t)(b * L_ + l0 + ll) * (DR + 2 * DS);
            sB[ll][s] = xp[r + DR + s];
            sC[ll][s] = xp[r + DR + DS + s];
        }
        __syncthreads();
#pragma unroll 4
        for (int lc = 0; lc < CHUNK; ++lc) {
            const int l = l0 + lc;
            const size_t e = (size_t)(b * L_ + l) * DI + d;
            float xv  = bf16_to_f32(xact[e]);
            float dm  = dmod[e];
            float dtv = clampf(squareplus3(dm) * 0.01f, 0.0f, 0.1f);
            float xdt = xv * dtv;
            float y = 0.0f;
#pragma unroll
            for (int s = 0; s < DS; ++s) {
                float u   = clampf(xdt * sB[lc][s], -100.0f, 100.0f);
                float dec = clampf(basev[s] + dm, 0.0f, 32000.0f) * rsh[s];
                h[s] = h[s] * dec + u;
                y += clampf(h[s], -1000.0f, 1000.0f) * sC[lc][s];
            }
            y = clampf(y, -100.0f, 100.0f);
            float z = xz[(size_t)(b * L_ + l) * (2 * DI) + DI + d];
            y *= 0.5f + 0.5f * z / (1.0f + fabsf(z));
            ybf[e] = f32_to_bf16(y);
        }
    }
}

// ---------- host ----------
extern "C" void kernel_launch(void* const* d_in, const int* in_sizes, int n_in,
                              void* d_out, int out_size, void* d_ws, size_t ws_size,
                              hipStream_t stream) {
    const float* hidden     = (const float*)d_in[0];
    const float* gamma      = (const float*)d_in[1];
    const float* W_in       = (const float*)d_in[2];
    const float* conv_w     = (const float*)d_in[3];
    const float* conv_b     = (const float*)d_in[4];
    const float* W_x        = (const float*)d_in[5];
    const float* W_dt       = (const float*)d_in[6];
    const float* W_out      = (const float*)d_in[7];
    const float* base_decay = (const float*)d_in[8];
    const float* res_gate   = (const float*)d_in[9];
    const float* dec_shift  = (const float*)d_in[10];
    float* out = (float*)d_out;

    char* ws = (char*)d_ws;
    size_t off = 0;
    auto alloc = [&](size_t bytes) -> void* {
        void* p = ws + off;
        off += (bytes + 255) & ~(size_t)255;
        return p;
    };
    const int MT = B_ * L_;                                   // 2048 token rows
    unsigned short* hsn    = (unsigned short*)alloc((size_t)MT * DM * 2);
    unsigned short* Win_b  = (unsigned short*)alloc((size_t)2 * DI * DM * 2);
    unsigned short* Wx_b   = (unsigned short*)alloc((size_t)(DR + 2 * DS) * DI * 2);
    unsigned short* Wdt_b  = (unsigned short*)alloc((size_t)DI * DR * 2);
    unsigned short* Wout_b = (unsigned short*)alloc((size_t)DM * DI * 2);
    float*          xz     = (float*)alloc((size_t)MT * 2 * DI * 4);
    unsigned short* xact   = (unsigned short*)alloc((size_t)MT * DI * 2);
    float*          xpm    = (float*)alloc((size_t)MT * (DR + 2 * DS) * 4);
    unsigned short* dtb    = (unsigned short*)alloc((size_t)MT * DR * 2);
    float*          dmod   = (float*)alloc((size_t)MT * DI * 4);
    unsigned short* ybf    = (unsigned short*)alloc((size_t)MT * DI * 2);
    (void)ws_size; (void)in_sizes; (void)n_in; (void)out_size;

    // weights -> bf16 (L2-resident thereafter: 12.5 MB << 192 MB)
    int nWin = 2 * DI * DM, nWx = (DR + 2 * DS) * DI, nWdt = DI * DR, nWo = DM * DI;
    cvt_bf16_kernel<<<(nWin + 255) / 256, 256, 0, stream>>>(W_in,  Win_b,  nWin);
    cvt_bf16_kernel<<<(nWx  + 255) / 256, 256, 0, stream>>>(W_x,   Wx_b,   nWx);
    cvt_bf16_kernel<<<(nWdt + 255) / 256, 256, 0, stream>>>(W_dt,  Wdt_b,  nWdt);
    cvt_bf16_kernel<<<(nWo  + 255) / 256, 256, 0, stream>>>(W_out, Wout_b, nWo);

    // 1) bitshift norm
    norm_kernel<<<MT, 256, 0, stream>>>(hidden, gamma, hsn);

    // 2) xz = clip(hs @ W_in^T)  : M=2048 N=4096 K=1024  (register-blocked)
    gemm_bf16_wmma_big<<<dim3(2 * DI / 64, MT / 128), 128, 0, stream>>>(
        hsn, Win_b, xz, nullptr, nullptr, MT, 2 * DI, DM, -100.0f, 100.0f);

    // 3) causal conv4 + squareplus
    conv_act_kernel<<<(B_ * L_ * DI + 255) / 256, 256, 0, stream>>>(xz, conv_w, conv_b, xact);

    // 4) xp = clip(x @ W_x^T)    : M=2048 N=96 K=2048  (small-N kernel)
    gemm_bf16_wmma<<<dim3((DR + 2 * DS + 63) / 64, MT / 16), 128, 0, stream>>>(
        xact, Wx_b, xpm, MT, DR + 2 * DS, DI, -100.0f, 100.0f);

    // 5) dt slice -> bf16
    slice_dt_kernel<<<(MT * DR + 255) / 256, 256, 0, stream>>>(xpm, dtb);

    // 6) decay_mod = clip(dt @ W_dt^T, +-20) : M=2048 N=2048 K=64 (register-blocked)
    gemm_bf16_wmma_big<<<dim3(DI / 64, MT / 128), 128, 0, stream>>>(
        dtb, Wdt_b, dmod, nullptr, nullptr, MT, DI, DR, -20.0f, 20.0f);

    // 7) fused scan + C contraction + gate (h never hits memory)
    scan_kernel<<<B_ * DI / 256, 256, 0, stream>>>(
        xact, dmod, xpm, xz, base_decay, dec_shift, ybf);

    // 8) out = residual + clip(y @ W_out^T) * res_gate : M=2048 N=1024 K=2048 (register-blocked)
    gemm_bf16_wmma_big<<<dim3(DM / 64, MT / 128), 128, 0, stream>>>(
        ybf, Wout_b, out, hidden, res_gate, MT, DM, DI, -100.0f, 100.0f);
}
